// AttnDecoderRNN_62551903699604
// MI455X (gfx1250) — matrix operand order repaired and checked by reading, not compiled, not run on hardware
//
#include <hip/hip_runtime.h>
#include <hip/hip_bf16.h>
#include <cstdint>

// ---------------- problem constants ----------------
constexpr int B_    = 64;
constexpr int S_    = 256;
constexpr int EH2_  = 1024;
constexpr int DEC_  = 1024;
constexpr int E_    = 512;
constexpr int NOTE_ = 128;
constexpr int DUR_  = 64;
constexpr int GAP_  = 64;
constexpr int T_    = 100;
constexpr int XDIM_ = 3 * E_ + EH2_;     // 2560 (GRU input)
constexpr int CAT_  = DEC_ + EH2_ + E_;  // 2560 (head input)
constexpr int G3_   = 3 * DEC_;          // 3072 (gates)

typedef __bf16 v16bf __attribute__((ext_vector_type(16)));
typedef float  v8f   __attribute__((ext_vector_type(8)));
typedef int    v4i_  __attribute__((vector_size(16)));   // 16B payload type for async LDS copy

// ---------------- async global->LDS staging (CDNA5) with safe fallback ----------------
#if defined(__has_builtin)
#if __has_builtin(__builtin_amdgcn_global_load_async_to_lds_b128) && \
    __has_builtin(__builtin_amdgcn_s_wait_asynccnt)
#define GEMM_ASYNC_COPY 1
#endif
#endif
#ifndef GEMM_ASYNC_COPY
#define GEMM_ASYNC_COPY 0
#endif

__device__ __forceinline__ void lds_copy16(const __bf16* g, __bf16* l) {
#if GEMM_ASYNC_COPY
  // AS1 (global) pointer keeps the flat address value; AS3 (LDS) pointer is the
  // low 32 bits of the flat LDS address.
  v4i_ __attribute__((address_space(1)))* gp =
      (v4i_ __attribute__((address_space(1)))*)(uintptr_t)g;
  v4i_ __attribute__((address_space(3)))* lp =
      (v4i_ __attribute__((address_space(3)))*)(uint32_t)(uintptr_t)l;
  __builtin_amdgcn_global_load_async_to_lds_b128(gp, lp, 0, 0);
#else
  *(uint4*)l = *(const uint4*)g;
#endif
}

__device__ __forceinline__ void wait_async_le4() {
#if GEMM_ASYNC_COPY
  __builtin_amdgcn_s_wait_asynccnt(4);
#endif
}
__device__ __forceinline__ void wait_async_le0() {
#if GEMM_ASYNC_COPY
  __builtin_amdgcn_s_wait_asynccnt(0);
#endif
}

// ---------------- generic fp32 -> bf16 convert ----------------
__global__ void convert_bf16_kernel(const float* __restrict__ src,
                                    __bf16* __restrict__ dst, long n) {
  long i = (long)blockIdx.x * blockDim.x + threadIdx.x;
  if (i < n) dst[i] = (__bf16)src[i];
}

__global__ void copy_f32_kernel(const float* __restrict__ src,
                                float* __restrict__ dst, long n) {
  long i = (long)blockIdx.x * blockDim.x + threadIdx.x;
  if (i < n) dst[i] = src[i];
}

// init h = encoder_hidden (fp32 + bf16 shadow)
__global__ void init_h_kernel(const float* __restrict__ src,
                              float* __restrict__ h,
                              __bf16* __restrict__ h_bf, long n) {
  long i = (long)blockIdx.x * blockDim.x + threadIdx.x;
  if (i < n) { float v = src[i]; h[i] = v; h_bf[i] = (__bf16)v; }
}

// ---------------- embedding gather: emb_bf[t][k][b][e] ----------------
__global__ void embed_kernel(const int* __restrict__ target,
                             const float* __restrict__ table,
                             __bf16* __restrict__ emb) {
  long idx = (long)blockIdx.x * blockDim.x + threadIdx.x;
  const long total = (long)T_ * 3 * B_ * E_;
  if (idx >= total) return;
  int e = (int)(idx % E_); long r = idx / E_;
  int b = (int)(r % B_);   r /= B_;
  int k = (int)(r % 3);
  int t = (int)(r / 3);
  int tok = (t == 0) ? 0 : target[((long)b * T_ + (t - 1)) * 3 + k]; // SOS=0
  emb[idx] = (__bf16)table[(long)tok * E_ + e];
}

// ---------------- LDS-tiled WMMA bf16 GEMM ----------------
// C[M,N] = A[M,K] * W[N,K]^T (+bias).
// Block = 128 threads (4 waves, 2x2), tile 64(M) x 64(N), K-staged by 32 with
// double-buffered LDS fed by async global->LDS b128 copies (ASYNCcnt).
// Each wave computes a 32x32 sub-tile as 2x2 WMMA accumulators.
// grid = (N/64, M/64). Requires M%64==0, N%64==0, K%32==0 (true for all calls).
constexpr int BM_ = 64, BN_ = 64, BK_ = 32;
constexpr int LSTR_ = 40;  // padded LDS row stride (elements): 80B, 16B-aligned

__global__ __launch_bounds__(128)
void gemm_bf16_wmma_kernel(const __bf16* __restrict__ A,
                           const __bf16* __restrict__ W,
                           const float* __restrict__ bias,
                           float* __restrict__ C,
                           int K, int lda, int ldw, int ldc) {
  __shared__ alignas(16) __bf16 Asm[2][BM_ * LSTR_];
  __shared__ alignas(16) __bf16 Bsm[2][BN_ * LSTR_];

  const int tid  = threadIdx.x;   // 0..127
  const int wave = tid >> 5;      // 0..3
  const int lane = tid & 31;
  const int half = lane >> 4;     // 0: lanes 0-15, 1: lanes 16-31
  const int l    = lane & 15;
  const int wm   = wave >> 1;     // M strip of 32 rows
  const int wn   = wave & 1;      // N strip of 32 cols

  const long m0 = (long)blockIdx.y * BM_;
  const long n0 = (long)blockIdx.x * BN_;

  const int nstages = K / BK_;

  // copy mapping: 64x32 bf16 tile = 256 chunks of 16B; 2 chunks/thread/tile.
  // chunk c: row = c>>2, col8 = (c&3)*8
  auto stage_copy = [&](int s) {
    const int buf = s & 1;
    const long kk = (long)s * BK_;
#pragma unroll
    for (int j = 0; j < 2; ++j) {
      const int c    = tid + j * 128;
      const int row  = c >> 2;
      const int col8 = (c & 3) * 8;
      lds_copy16(A + (m0 + row) * (long)lda + kk + col8,
                 &Asm[buf][row * LSTR_ + col8]);
      lds_copy16(W + (n0 + row) * (long)ldw + kk + col8,
                 &Bsm[buf][row * LSTR_ + col8]);
    }
  };

  v8f acc[2][2];
#pragma unroll
  for (int mi = 0; mi < 2; ++mi)
#pragma unroll
    for (int ni = 0; ni < 2; ++ni)
      acc[mi][ni] = (v8f){0.f, 0.f, 0.f, 0.f, 0.f, 0.f, 0.f, 0.f};

  stage_copy(0);

  for (int s = 0; s < nstages; ++s) {
    if (s + 1 < nstages) {
      stage_copy(s + 1);     // prefetch next stage into other buffer
      wait_async_le4();      // release this wave's 4 copies for stage s
    } else {
      wait_async_le0();
    }
    __syncthreads();

    const __bf16* As = Asm[s & 1];
    const __bf16* Bs = Bsm[s & 1];

    union F { v16bf v; uint4 q[2]; };
    F fa[2], fb[2];
#pragma unroll
    for (int mi = 0; mi < 2; ++mi) {
      // A 16x32 frag: lanes 0-15 hold K [0..7],[16..23]; lanes 16-31 shifted +8
      const __bf16* ap = As + (wm * 32 + mi * 16 + l) * LSTR_ + half * 8;
      fa[mi].q[0] = *(const uint4*)ap;
      fa[mi].q[1] = *(const uint4*)(ap + 16);
    }
#pragma unroll
    for (int ni = 0; ni < 2; ++ni) {
      // B 32x16 frag from row-major W: lane's column n=l, K [0..7],[16..23]; lanes 16-31 shifted +8
      const __bf16* bp = Bs + (wn * 32 + ni * 16 + l) * LSTR_ + half * 8;
      fb[ni].q[0] = *(const uint4*)bp;
      fb[ni].q[1] = *(const uint4*)(bp + 16);
    }
#pragma unroll
    for (int mi = 0; mi < 2; ++mi)
#pragma unroll
      for (int ni = 0; ni < 2; ++ni)
        acc[mi][ni] = __builtin_amdgcn_wmma_f32_16x16x32_bf16(
            false, fa[mi].v, false, fb[ni].v, (short)0, acc[mi][ni], false, false);

    __syncthreads();
  }

  // epilogue: C/D layout -> VGPR r holds (M = half*8 + r, N = l)
#pragma unroll
  for (int ni = 0; ni < 2; ++ni) {
    const long ncol = n0 + wn * 32 + ni * 16 + l;
    const float bval = bias ? bias[ncol] : 0.0f;
#pragma unroll
    for (int mi = 0; mi < 2; ++mi) {
      float* Cp = C + (m0 + wm * 32 + mi * 16 + half * 8) * (long)ldc + ncol;
#pragma unroll
      for (int r = 0; r < 8; ++r) Cp[(long)r * ldc] = acc[mi][ni][r] + bval;
    }
  }
}

// ---------------- attention: score = v . tanh(enc_proj + q); softmax over S ----------------
__global__ void attn_softmax_kernel(const float* __restrict__ enc_proj,
                                    const float* __restrict__ q,
                                    const float* __restrict__ vvec,
                                    float* __restrict__ attn,
                                    float* __restrict__ out_attns,
                                    int t) {
  __shared__ float qs[DEC_];
  __shared__ float vs[DEC_];
  __shared__ float red[S_];
  const int b = blockIdx.x;
  const int s = threadIdx.x;  // 256 threads = S_
  for (int d = s; d < DEC_; d += S_) { qs[d] = q[(long)b * DEC_ + d]; vs[d] = vvec[d]; }
  __syncthreads();
  const float* ep = enc_proj + ((long)b * S_ + s) * DEC_;
  float acc = 0.f;
#pragma unroll 4
  for (int d = 0; d < DEC_; ++d) acc += vs[d] * tanhf(ep[d] + qs[d]);
  red[s] = acc; __syncthreads();
  for (int st = S_ / 2; st > 0; st >>= 1) {
    if (s < st) red[s] = fmaxf(red[s], red[s + st]);
    __syncthreads();
  }
  const float m = red[0]; __syncthreads();
  const float e = expf(acc - m);
  red[s] = e; __syncthreads();
  for (int st = S_ / 2; st > 0; st >>= 1) {
    if (s < st) red[s] += red[s + st];
    __syncthreads();
  }
  const float a = e / red[0];
  attn[(long)b * S_ + s] = a;
  out_attns[((long)b * T_ + t) * S_ + s] = a;
}

// ---------------- weighted[b,f] = sum_s attn[b,s] * enc_out[b,s,f] ----------------
__global__ void weighted_kernel(const float* __restrict__ attn,
                                const float* __restrict__ enc_out,
                                float* __restrict__ weighted,
                                __bf16* __restrict__ weighted_bf) {
  const int b = blockIdx.y;
  const int f = blockIdx.x * blockDim.x + threadIdx.x;
  const float* eb = enc_out + (long)b * S_ * EH2_ + f;
  const float* ab = attn + (long)b * S_;
  float acc = 0.f;
#pragma unroll 4
  for (int s = 0; s < S_; ++s) acc += ab[s] * eb[(long)s * EH2_];
  weighted[(long)b * EH2_ + f] = acc;
  weighted_bf[(long)b * EH2_ + f] = (__bf16)acc;
}

// ---------------- x = [emb0, emb1, emb2, weighted] (bf16) ----------------
__global__ void build_x_kernel(const __bf16* __restrict__ emb_t,  // [3][B][E]
                               const __bf16* __restrict__ wbf,    // [B][EH2]
                               __bf16* __restrict__ x) {          // [B][XDIM]
  const int b = blockIdx.y;
  const int i = blockIdx.x * blockDim.x + threadIdx.x;
  __bf16 v;
  if (i < 3 * E_) {
    const int k = i / E_, e = i % E_;
    v = emb_t[((long)k * B_ + b) * E_ + e];
  } else {
    v = wbf[(long)b * EH2_ + (i - 3 * E_)];
  }
  x[(long)b * XDIM_ + i] = v;
}

// ---------------- GRU elementwise ----------------
__global__ void gru_kernel(const float* __restrict__ gi,  // [B][3*DEC] x-part (+b_ih)
                           const float* __restrict__ gh,  // [B][3*DEC] h-part (+b_hh)
                           const float* __restrict__ h,
                           float* __restrict__ h_new,
                           __bf16* __restrict__ h_new_bf) {
  const int b = blockIdx.y;
  const int d = blockIdx.x * blockDim.x + threadIdx.x;
  const long o = (long)b * G3_;
  const float ir = gi[o + d], iz = gi[o + DEC_ + d], in = gi[o + 2 * DEC_ + d];
  const float hr = gh[o + d], hz = gh[o + DEC_ + d], hn = gh[o + 2 * DEC_ + d];
  const float r = 1.f / (1.f + expf(-(ir + hr)));
  const float z = 1.f / (1.f + expf(-(iz + hz)));
  const float n = tanhf(in + r * hn);
  const float hv = h[(long)b * DEC_ + d];
  const float out = (1.f - z) * n + z * hv;
  h_new[(long)b * DEC_ + d] = out;
  h_new_bf[(long)b * DEC_ + d] = (__bf16)out;
}

// ---------------- head concat: cat[k][b] = [h_new, weighted, emb_k] ----------------
__global__ void build_cats_kernel(const __bf16* __restrict__ h_bf,
                                  const __bf16* __restrict__ wbf,
                                  const __bf16* __restrict__ emb_t,  // [3][B][E]
                                  __bf16* __restrict__ cat) {        // [3][B][CAT_]
  const int k = blockIdx.z;
  const int b = blockIdx.y;
  const int i = blockIdx.x * blockDim.x + threadIdx.x;
  __bf16 v;
  if (i < DEC_)             v = h_bf[(long)b * DEC_ + i];
  else if (i < DEC_ + EH2_) v = wbf[(long)b * EH2_ + (i - DEC_)];
  else                      v = emb_t[((long)k * B_ + b) * E_ + (i - DEC_ - EH2_)];
  cat[((long)k * B_ + b) * CAT_ + i] = v;
}

// ---------------- in-place log_softmax over last dim ----------------
__global__ void log_softmax_kernel(float* __restrict__ data, int ncls) {
  __shared__ float sm[128];
  const long row = blockIdx.x;
  float* p = data + row * (long)ncls;
  const int i = threadIdx.x;
  const float x = p[i];
  sm[i] = x; __syncthreads();
  for (int st = blockDim.x / 2; st > 0; st >>= 1) {
    if (i < st) sm[i] = fmaxf(sm[i], sm[i + st]);
    __syncthreads();
  }
  const float m = sm[0]; __syncthreads();
  const float e = expf(x - m);
  sm[i] = e; __syncthreads();
  for (int st = blockDim.x / 2; st > 0; st >>= 1) {
    if (i < st) sm[i] += sm[i + st];
    __syncthreads();
  }
  p[i] = x - m - logf(sm[0]);
}

// =======================================================================
extern "C" void kernel_launch(void* const* d_in, const int* in_sizes, int n_in,
                              void* d_out, int out_size, void* d_ws, size_t ws_size,
                              hipStream_t stream) {
  const float* enc_out   = (const float*)d_in[0];
  const float* enc_hid   = (const float*)d_in[1];
  const int*   target    = (const int*)d_in[2];
  const float* emb_table = (const float*)d_in[3];
  const float* Wa        = (const float*)d_in[4];
  const float* ba        = (const float*)d_in[5];
  const float* vvec      = (const float*)d_in[6];
  const float* W_ih      = (const float*)d_in[7];
  const float* W_hh      = (const float*)d_in[8];
  const float* b_ih      = (const float*)d_in[9];
  const float* b_hh      = (const float*)d_in[10];
  const float* Wn        = (const float*)d_in[11];
  const float* bn        = (const float*)d_in[12];
  const float* Wd        = (const float*)d_in[13];
  const float* bd        = (const float*)d_in[14];
  const float* Wg        = (const float*)d_in[15];
  const float* bg        = (const float*)d_in[16];

  float* out = (float*)d_out;
  // output layout (flat, return order)
  float* out_notes = out;                                              // B*T*NOTE
  float* out_durs  = out_notes + (long)B_ * T_ * NOTE_;                // B*T*DUR
  float* out_gaps  = out_durs  + (long)B_ * T_ * DUR_;                 // B*T*GAP
  float* out_h     = out_gaps  + (long)B_ * T_ * GAP_;                 // B*DEC
  float* out_attns = out_h     + (long)B_ * DEC_;                      // B*T*S

  // -------- workspace bump allocator --------
  char* ws = (char*)d_ws;
  size_t off = 0;
  auto alloc = [&](size_t bytes) -> void* {
    off = (off + 255) & ~(size_t)255;
    void* p = ws + off;
    off += bytes;
    return p;
  };
  __bf16* enc_out_bf = (__bf16*)alloc((size_t)B_ * S_ * EH2_ * 2);
  __bf16* Wa_bf      = (__bf16*)alloc((size_t)DEC_ * (DEC_ + EH2_) * 2);
  __bf16* Wih_bf     = (__bf16*)alloc((size_t)G3_ * XDIM_ * 2);
  __bf16* Whh_bf     = (__bf16*)alloc((size_t)G3_ * DEC_ * 2);
  __bf16* Wn_bf      = (__bf16*)alloc((size_t)NOTE_ * CAT_ * 2);
  __bf16* Wd_bf      = (__bf16*)alloc((size_t)DUR_ * CAT_ * 2);
  __bf16* Wg_bf      = (__bf16*)alloc((size_t)GAP_ * CAT_ * 2);
  __bf16* emb_bf     = (__bf16*)alloc((size_t)T_ * 3 * B_ * E_ * 2);
  float*  enc_proj   = (float*)alloc((size_t)B_ * S_ * DEC_ * 4);
  float*  qbuf       = (float*)alloc((size_t)B_ * DEC_ * 4);
  float*  attn       = (float*)alloc((size_t)B_ * S_ * 4);
  float*  weighted   = (float*)alloc((size_t)B_ * EH2_ * 4);
  __bf16* wgt_bf     = (__bf16*)alloc((size_t)B_ * EH2_ * 2);
  __bf16* x_bf       = (__bf16*)alloc((size_t)B_ * XDIM_ * 2);
  float*  gi         = (float*)alloc((size_t)B_ * G3_ * 4);
  float*  gh         = (float*)alloc((size_t)B_ * G3_ * 4);
  float*  hA         = (float*)alloc((size_t)B_ * DEC_ * 4);
  float*  hB         = (float*)alloc((size_t)B_ * DEC_ * 4);
  __bf16* h_bf       = (__bf16*)alloc((size_t)B_ * DEC_ * 2);
  __bf16* cats       = (__bf16*)alloc((size_t)3 * B_ * CAT_ * 2);
  (void)ws_size; (void)n_in; (void)in_sizes; (void)out_size;

  auto cvt = [&](const float* s, __bf16* d, long n) {
    convert_bf16_kernel<<<dim3((unsigned)((n + 255) / 256)), 256, 0, stream>>>(s, d, n);
  };
  auto gemm = [&](const __bf16* A, const __bf16* W, const float* bias, float* C,
                  int M, int N, int K, int lda, int ldw, int ldc) {
    gemm_bf16_wmma_kernel<<<dim3(N / BN_, M / BM_), 128, 0, stream>>>(
        A, W, bias, C, K, lda, ldw, ldc);
  };

  // -------- precompute --------
  cvt(enc_out, enc_out_bf, (long)B_ * S_ * EH2_);
  cvt(Wa,      Wa_bf,      (long)DEC_ * (DEC_ + EH2_));
  cvt(W_ih,    Wih_bf,     (long)G3_ * XDIM_);
  cvt(W_hh,    Whh_bf,     (long)G3_ * DEC_);
  cvt(Wn,      Wn_bf,      (long)NOTE_ * CAT_);
  cvt(Wd,      Wd_bf,      (long)DUR_ * CAT_);
  cvt(Wg,      Wg_bf,      (long)GAP_ * CAT_);

  {
    const long n = (long)T_ * 3 * B_ * E_;
    embed_kernel<<<dim3((unsigned)((n + 255) / 256)), 256, 0, stream>>>(target, emb_table, emb_bf);
  }
  init_h_kernel<<<dim3((B_ * DEC_ + 255) / 256), 256, 0, stream>>>(enc_hid, hA, h_bf, (long)B_ * DEC_);

  // enc_proj = enc_out @ Wa_e^T + ba   (M=B*S, N=DEC, K=EH2); Wa_e rows start at col DEC
  gemm(enc_out_bf, Wa_bf + DEC_, ba, enc_proj,
       B_ * S_, DEC_, EH2_, EH2_, DEC_ + EH2_, DEC_);

  float* h_cur  = hA;
  float* h_next = hB;

  // -------- sequential scan --------
  for (int t = 0; t < T_; ++t) {
    const __bf16* emb_t = emb_bf + (long)t * 3 * B_ * E_;

    // q = h @ Wa_h^T  (M=B, N=DEC, K=DEC); Wa_h rows at col 0, row stride DEC+EH2
    gemm(h_bf, Wa_bf, nullptr, qbuf, B_, DEC_, DEC_, DEC_, DEC_ + EH2_, DEC_);

    attn_softmax_kernel<<<dim3(B_), S_, 0, stream>>>(enc_proj, qbuf, vvec, attn, out_attns, t);

    weighted_kernel<<<dim3(EH2_ / 256, B_), 256, 0, stream>>>(attn, enc_out, weighted, wgt_bf);

    build_x_kernel<<<dim3(XDIM_ / 256, B_), 256, 0, stream>>>(emb_t, wgt_bf, x_bf);

    // gi = x @ W_ih^T + b_ih   (M=B, N=3*DEC, K=XDIM)
    gemm(x_bf, Wih_bf, b_ih, gi, B_, G3_, XDIM_, XDIM_, XDIM_, G3_);
    // gh = h @ W_hh^T + b_hh   (M=B, N=3*DEC, K=DEC)
    gemm(h_bf, Whh_bf, b_hh, gh, B_, G3_, DEC_, DEC_, DEC_, G3_);

    gru_kernel<<<dim3(DEC_ / 256, B_), 256, 0, stream>>>(gi, gh, h_cur, h_next, h_bf);

    build_cats_kernel<<<dim3(CAT_ / 256, B_, 3), 256, 0, stream>>>(h_bf, wgt_bf, emb_t, cats);

    // heads -> write logits straight into output layout [b][t][n] (ldc = T*ncls)
    gemm(cats + 0L * B_ * CAT_, Wn_bf, bn, out_notes + (long)t * NOTE_,
         B_, NOTE_, CAT_, CAT_, CAT_, T_ * NOTE_);
    gemm(cats + 1L * B_ * CAT_, Wd_bf, bd, out_durs + (long)t * DUR_,
         B_, DUR_, CAT_, CAT_, CAT_, T_ * DUR_);
    gemm(cats + 2L * B_ * CAT_, Wg_bf, bg, out_gaps + (long)t * GAP_,
         B_, GAP_, CAT_, CAT_, CAT_, T_ * GAP_);

    // swap h buffers
    float* tmp = h_cur; h_cur = h_next; h_next = tmp;
  }

  // h_final
  copy_f32_kernel<<<dim3((B_ * DEC_ + 255) / 256), 256, 0, stream>>>(h_cur, out_h, (long)B_ * DEC_);

  // log_softmax over class dims
  log_softmax_kernel<<<dim3(B_ * T_), NOTE_, 0, stream>>>(out_notes, NOTE_);
  log_softmax_kernel<<<dim3(B_ * T_), DUR_,  0, stream>>>(out_durs,  DUR_);
  log_softmax_kernel<<<dim3(B_ * T_), GAP_,  0, stream>>>(out_gaps,  GAP_);
}
